// HierarchicalGroupStageMoE_24446953849154
// MI455X (gfx1250) — matrix operand order repaired and checked by reading, not compiled, not run on hardware
//
#include <hip/hip_runtime.h>
#include <hip/hip_bf16.h>

typedef __attribute__((ext_vector_type(16))) __bf16        v16bf;
typedef __attribute__((ext_vector_type(8)))  float         v8f;
typedef __attribute__((ext_vector_type(8)))  unsigned int  v8u;

namespace {
constexpr int Bb = 8, Ssq = 2048, Tt = Bb * Ssq;   // 16384 tokens
constexpr int Dm = 512, Ff = 64, DFe = 64, DHh = 256, DRr = 128;
constexpr int Gg = 8, ESs = 2, Ee = Gg * ESs;      // 16 experts
constexpr int RIN = Dm + DFe;                      // 576
// LDS row strides padded +4 dwords: lanes land on distinct 4-bank groups
// (4*mrow mod 64), killing the 16-way conflict of power-of-2 strides, while
// keeping every row base 16B-aligned (260*4=65*16, 132*4=33*16, 36*4=9*16).
constexpr int ROWH = 260;   // H tile row stride in dwords (512 bf16 + pad)
constexpr int ROWY = 132;   // Y tile row stride in dwords (256 bf16 + pad)
constexpr int ROWS = 36;    // stage/gemb tile row stride in dwords (64 bf16 + pad)
}

__device__ __forceinline__ unsigned short f2bf(float f) {
  unsigned u = __builtin_bit_cast(unsigned, f);
  u += 0x7fffu + ((u >> 16) & 1u);                 // round-to-nearest-even
  return (unsigned short)(u >> 16);
}

__device__ __forceinline__ float geluf(float x) {
  return 0.5f * x * (1.0f + erff(x * 0.70710678118654752f));
}

__device__ __forceinline__ v8f wmma_bf16(const v8u& a, const v8u& b, v8f c) {
  return __builtin_amdgcn_wmma_f32_16x16x32_bf16(
      false, __builtin_bit_cast(v16bf, a),
      false, __builtin_bit_cast(v16bf, b),
      (short)0, c, false, false);
}

// A-fragment (16-bit, 16x32): two ds_load_b128 from rowp = row + kt*16 + half*4.
__device__ __forceinline__ v8u load_a_frag(const unsigned* rowp) {
  const uint4* ap = (const uint4*)rowp;
  uint4 lo = ap[0], hi = ap[2];
  v8u a;
  a[0] = lo.x; a[1] = lo.y; a[2] = lo.z; a[3] = lo.w;
  a[4] = hi.x; a[5] = hi.y; a[6] = hi.z; a[7] = hi.w;
  return a;
}

// B-fragment from fragment-native packed weights [kt][n][16 dwords]:
// lane needs dwords half*4 + p and half*4 + 8 + p -> two b128 loads.
__device__ __forceinline__ v8u load_b_frag(const unsigned* __restrict__ Wp,
                                           int kt, int N, int n, int half) {
  const uint4* bp = (const uint4*)(Wp + (((size_t)(kt * N + n)) << 4) + half * 4);
  uint4 lo = bp[0], hi = bp[2];
  v8u b;
  b[0] = lo.x; b[1] = lo.y; b[2] = lo.z; b[3] = lo.w;
  b[4] = hi.x; b[5] = hi.y; b[6] = hi.z; b[7] = hi.w;
  return b;
}

// ---------------------------------------------------------------------------
// Pack fp32 blocks [nblk][Kg][N] -> per block: [Kg/32][N][16] dwords, where
// dword w of tile (kt,n) holds bf16 pair {W[2*(kt*16+w)][n], W[2*(kt*16+w)+1][n]}.
// ---------------------------------------------------------------------------
__global__ void pack_b_frag(const float* __restrict__ W, unsigned* __restrict__ out,
                            int Kg, int N, int perblk, int total) {
  int i = blockIdx.x * 256 + threadIdx.x;
  if (i >= total) return;
  int blk = i / perblk, o = i - blk * perblk;
  int w = o & 15, r = o >> 4;
  int n = r % N, kt = r / N;
  int k2 = kt * 16 + w;
  const float* Wb = W + (size_t)blk * Kg * N;
  unsigned lo = f2bf(Wb[(size_t)(2 * k2) * N + n]);
  unsigned hi = f2bf(Wb[(size_t)(2 * k2 + 1) * N + n]);
  out[i] = lo | (hi << 16);
}

// ---------------------------------------------------------------------------
// Per-token prep: LayerNorm -> h(bf16), stage_emb(bf16), per-group gemb(bf16)
// ---------------------------------------------------------------------------
__global__ __launch_bounds__(256) void prep_kernel(
    const float* __restrict__ hidden, const float* __restrict__ feat,
    const float* __restrict__ ln_g, const float* __restrict__ ln_b,
    const float* __restrict__ stageW, const float* __restrict__ stageb,
    const float* __restrict__ gfW, const float* __restrict__ gfb,
    unsigned short* __restrict__ h_bf, unsigned short* __restrict__ stage_bf,
    unsigned short* __restrict__ gemb_bf) {
  __shared__ float red[256];
  __shared__ float featl[Ff];
  __shared__ float stats[2];
  int t = blockIdx.x, tid = threadIdx.x;
  const float* x = hidden + (size_t)t * Dm;
  float a0 = x[tid], a1 = x[tid + 256];
  red[tid] = a0 + a1;
  __syncthreads();
  for (int s = 128; s > 0; s >>= 1) { if (tid < s) red[tid] += red[tid + s]; __syncthreads(); }
  if (tid == 0) stats[0] = red[0] * (1.0f / Dm);
  __syncthreads();
  float mu = stats[0];
  float d0 = a0 - mu, d1 = a1 - mu;
  red[tid] = d0 * d0 + d1 * d1;
  __syncthreads();
  for (int s = 128; s > 0; s >>= 1) { if (tid < s) red[tid] += red[tid + s]; __syncthreads(); }
  if (tid == 0) stats[1] = rsqrtf(red[0] * (1.0f / Dm) + 1e-5f);
  if (tid < Ff) featl[tid] = feat[(size_t)t * Ff + tid];
  __syncthreads();
  float rstd = stats[1];
  h_bf[(size_t)t * Dm + tid]       = f2bf(d0 * rstd * ln_g[tid] + ln_b[tid]);
  h_bf[(size_t)t * Dm + tid + 256] = f2bf(d1 * rstd * ln_g[tid + 256] + ln_b[tid + 256]);
  if (tid < DFe) {
    float s = stageb[tid];
#pragma unroll 8
    for (int f = 0; f < Ff; ++f) s += featl[f] * stageW[f * DFe + tid];
    stage_bf[(size_t)t * DFe + tid] = f2bf(s);
  }
  for (int i = tid; i < Gg * DFe; i += 256) {
    int g = i >> 6, n = i & 63;
    float s = gfb[i];
#pragma unroll
    for (int f = 0; f < 8; ++f) s += featl[g * 8 + f] * gfW[(g * 8 + f) * DFe + n];
    gemb_bf[(size_t)t * (Gg * DFe) + i] = f2bf(s);
  }
}

// ---------------------------------------------------------------------------
// Router kernel: one block per (16-token tile, group g).
// ---------------------------------------------------------------------------
__global__ __launch_bounds__(256) void router_kernel(
    const unsigned short* __restrict__ h_bf, const unsigned short* __restrict__ stage_bf,
    const unsigned short* __restrict__ gemb_bf,
    const unsigned* __restrict__ psW1p, const float* __restrict__ psb1,
    const float* __restrict__ psW2, const float* __restrict__ psb2,
    const unsigned* __restrict__ irW1p, const float* __restrict__ irb1,
    const float* __restrict__ irW2, const float* __restrict__ irb2,
    float* __restrict__ glog, float* __restrict__ ilog) {
  __shared__ __align__(16) unsigned lds_h[16 * ROWH];
  __shared__ __align__(16) unsigned lds_sa[16 * ROWS];
  __shared__ __align__(16) unsigned lds_ga[16 * ROWS];
  __shared__ float lds_gh[16 * DRr];
  __shared__ float lds_ih[16 * DRr];
  int tile = blockIdx.x, g = blockIdx.y;
  int t0 = tile * 16, tid = threadIdx.x, lane = tid & 31, wv = tid >> 5;
  const uint4* hg = (const uint4*)(h_bf + (size_t)t0 * Dm);
  for (int i = tid; i < 16 * 64; i += 256) {
    int r = i >> 6, c = i & 63;
    ((uint4*)(lds_h + r * ROWH))[c] = hg[i];
  }
  const uint4* sg = (const uint4*)(stage_bf + (size_t)t0 * DFe);
  for (int i = tid; i < 16 * 8; i += 256) {
    int r = i >> 3, c = i & 7;
    ((uint4*)(lds_sa + r * ROWS))[c] = sg[i];
  }
  for (int i = tid; i < 16 * 8; i += 256) {
    int r = i >> 3, c = i & 7;
    const uint4* gp = (const uint4*)(gemb_bf + ((size_t)(t0 + r) * Gg + g) * DFe);
    ((uint4*)(lds_ga + r * ROWS))[c] = gp[c];
  }
  __syncthreads();

  int half = lane >> 4, mrow = lane & 15, ncol = lane & 15;
  int n = wv * 16 + ncol;
  const unsigned* Wg = psW1p + (size_t)g * (RIN / 2) * DRr;
  const unsigned* Wi = irW1p + (size_t)g * (RIN / 2) * DRr;
  v8f accg = {0, 0, 0, 0, 0, 0, 0, 0};
  v8f acci = accg;
  for (int kt = 0; kt < 16; ++kt) {               // shared K over h (512)
    v8u au = load_a_frag(lds_h + mrow * ROWH + kt * 16 + half * 4);
    v8u bg = load_b_frag(Wg, kt, DRr, n, half);
    v8u bi = load_b_frag(Wi, kt, DRr, n, half);
    accg = wmma_bf16(au, bg, accg);
    acci = wmma_bf16(au, bi, acci);
  }
#pragma unroll
  for (int tt = 0; tt < 2; ++tt) {                // K tail (feature embeddings)
    v8u as = load_a_frag(lds_sa + mrow * ROWS + tt * 16 + half * 4);
    v8u ag = load_a_frag(lds_ga + mrow * ROWS + tt * 16 + half * 4);
    v8u bg = load_b_frag(Wg, 16 + tt, DRr, n, half);
    v8u bi = load_b_frag(Wi, 16 + tt, DRr, n, half);
    accg = wmma_bf16(as, bg, accg);
    acci = wmma_bf16(ag, bi, acci);
  }
#pragma unroll
  for (int r = 0; r < 8; ++r) {                   // C layout: M = r + half*8
    int m = r + half * 8;
    lds_gh[m * DRr + n] = geluf(accg[r] + psb1[g * DRr + n]);
    lds_ih[m * DRr + n] = geluf(acci[r] + irb1[g * DRr + n]);
  }
  __syncthreads();
  if (tid < 48) {                                 // 16 tokens x {glog, ilog0, ilog1}
    int kind = tid >> 4, m = tid & 15;
    if (kind == 0) {
      float s = psb2[g];
      for (int hh = 0; hh < DRr; ++hh) s += lds_gh[m * DRr + hh] * psW2[g * DRr + hh];
      glog[(size_t)(t0 + m) * Gg + g] = s;
    } else {
      int es = kind - 1;
      float s = irb2[g * ESs + es];
      for (int hh = 0; hh < DRr; ++hh) s += lds_ih[m * DRr + hh] * irW2[(g * DRr + hh) * ESs + es];
      ilog[((size_t)(t0 + m) * Gg + g) * ESs + es] = s;
    }
  }
}

// ---------------------------------------------------------------------------
// Routing combine: top-2 softmax over groups x top-1 over experts -> w[T,16]
// ---------------------------------------------------------------------------
__global__ void combine_kernel(const float* __restrict__ glog,
                               const float* __restrict__ ilog,
                               float* __restrict__ w) {
  int t = blockIdx.x * 256 + threadIdx.x;
  if (t >= Tt) return;
  float gl[Gg];
#pragma unroll
  for (int g = 0; g < Gg; ++g) gl[g] = glog[(size_t)t * Gg + g];
  int i1 = 0; float v1 = gl[0];
#pragma unroll
  for (int g = 1; g < Gg; ++g) if (gl[g] > v1) { v1 = gl[g]; i1 = g; }
  int i2 = -1; float v2 = -3.4e38f;
#pragma unroll
  for (int g = 0; g < Gg; ++g) if (g != i1 && gl[g] > v2) { v2 = gl[g]; i2 = g; }
  float e2 = __expf(v2 - v1);
  float p1 = 1.0f / (1.0f + e2), p2 = e2 / (1.0f + e2);
#pragma unroll
  for (int g = 0; g < Gg; ++g) {
    float gw = (g == i1) ? p1 : ((g == i2) ? p2 : 0.0f);
    float l0 = ilog[((size_t)t * Gg + g) * ESs + 0];
    float l1 = ilog[((size_t)t * Gg + g) * ESs + 1];
    float w0 = (l0 >= l1) ? 1.0f : 0.0f;          // top-1 softmax == winner 1.0
    w[(size_t)t * Ee + g * ESs + 0] = gw * w0;
    w[(size_t)t * Ee + g * ESs + 1] = gw * (1.0f - w0);
  }
}

// ---------------------------------------------------------------------------
// Fused dense-expert kernel: one block per 32-token tile (2 M-tiles), 8 waves.
// Each B-fragment feeds two WMMAs (M-tile reuse halves L2 weight traffic).
// Per expert e: Y = gelu(H @ W1_e + b1) scaled per-row by w[t,e], then
// out_acc += Y @ W2_e (weighted combine folded into GEMM2 accumulation).
// ---------------------------------------------------------------------------
__global__ __launch_bounds__(256) void expert_kernel(
    const unsigned short* __restrict__ h_bf,
    const unsigned* __restrict__ eW1p, const unsigned* __restrict__ eW2p,
    const float* __restrict__ eb1, const float* __restrict__ eb2,
    const float* __restrict__ wcomb, const float* __restrict__ hidden,
    const float* __restrict__ alpha, float* __restrict__ out) {
  __shared__ __align__(16) unsigned lds_h[32 * ROWH];  // 32.5 KB: H tile
  __shared__ __align__(16) unsigned lds_y[32 * ROWY];  // 16.5 KB: Y tile
  __shared__ float lds_w[32 * Ee];                     //  2 KB
  __shared__ __align__(16) float lds_eb2[Ee * Dm];     // 32 KB
  int t0 = blockIdx.x * 32, tid = threadIdx.x, lane = tid & 31, wv = tid >> 5;
  const uint4* hg = (const uint4*)(h_bf + (size_t)t0 * Dm);
  for (int i = tid; i < 32 * 64; i += 256) {
    int r = i >> 6, c = i & 63;
    ((uint4*)(lds_h + r * ROWH))[c] = hg[i];
  }
  lds_w[tid] = wcomb[(size_t)t0 * Ee + tid];
  lds_w[tid + 256] = wcomb[(size_t)t0 * Ee + tid + 256];
  for (int i = tid; i < Ee * Dm / 4; i += 256) ((uint4*)lds_eb2)[i] = ((const uint4*)eb2)[i];
  __syncthreads();

  int half = lane >> 4, mrow = lane & 15, ncol = lane & 15;
  v8f zero = {0, 0, 0, 0, 0, 0, 0, 0};
  v8f oacc[2][4] = {{zero, zero, zero, zero}, {zero, zero, zero, zero}};
  unsigned short* ys = (unsigned short*)lds_y;

  for (int e = 0; e < Ee; ++e) {
    const unsigned* W1 = eW1p + (size_t)e * (Dm / 2) * DHh;   // 65536 dwords
    const unsigned* W2 = eW2p + (size_t)e * (DHh / 2) * Dm;   // 65536 dwords
    if (e + 1 < Ee)
      __builtin_prefetch(eW1p + (size_t)(e + 1) * (Dm / 2) * DHh + tid * 16, 0, 1);
    // --- GEMM1: Y (32x256), wave owns cols [wv*32, wv*32+32) ---
    v8f yacc[2][2] = {{zero, zero}, {zero, zero}};
    for (int kt = 0; kt < 16; ++kt) {
      v8u a0 = load_a_frag(lds_h + mrow * ROWH + kt * 16 + half * 4);
      v8u a1 = load_a_frag(lds_h + (mrow + 16) * ROWH + kt * 16 + half * 4);
#pragma unroll
      for (int j = 0; j < 2; ++j) {
        int n = wv * 32 + j * 16 + ncol;
        v8u bu = load_b_frag(W1, kt, DHh, n, half);
        yacc[0][j] = wmma_bf16(a0, bu, yacc[0][j]);
        yacc[1][j] = wmma_bf16(a1, bu, yacc[1][j]);
      }
    }
    __syncthreads();   // previous expert's GEMM2 finished reading lds_y
#pragma unroll
    for (int mt = 0; mt < 2; ++mt) {
#pragma unroll
      for (int j = 0; j < 2; ++j) {
        int n = wv * 32 + j * 16 + ncol;
        float b1 = eb1[e * DHh + n];
#pragma unroll
        for (int r = 0; r < 8; ++r) {
          int m = mt * 16 + r + half * 8;
          float yv = geluf(yacc[mt][j][r] + b1) * lds_w[m * Ee + e];
          ys[m * (2 * ROWY) + n] = f2bf(yv);      // row stride = ROWY dwords
        }
      }
    }
    __syncthreads();
    // --- GEMM2: out_acc += Y @ W2_e, wave cols [wv*64, wv*64+64) ---
    for (int kt = 0; kt < 8; ++kt) {
      v8u a0 = load_a_frag(lds_y + mrow * ROWY + kt * 16 + half * 4);
      v8u a1 = load_a_frag(lds_y + (mrow + 16) * ROWY + kt * 16 + half * 4);
#pragma unroll
      for (int j = 0; j < 4; ++j) {
        int n = wv * 64 + j * 16 + ncol;
        v8u bu = load_b_frag(W2, kt, Dm, n, half);
        oacc[0][j] = wmma_bf16(a0, bu, oacc[0][j]);
        oacc[1][j] = wmma_bf16(a1, bu, oacc[1][j]);
      }
    }
  }
  __syncthreads();
  float av = alpha[0];
#pragma unroll
  for (int mt = 0; mt < 2; ++mt) {
#pragma unroll
    for (int r = 0; r < 8; ++r) {
      int m = mt * 16 + r + half * 8;
      float wrow[Ee];
#pragma unroll
      for (int e = 0; e < Ee; ++e) wrow[e] = lds_w[m * Ee + e];
#pragma unroll
      for (int j = 0; j < 4; ++j) {
        int d = wv * 64 + j * 16 + ncol;
        float bsum = 0.0f;
#pragma unroll
        for (int e = 0; e < Ee; ++e) bsum += wrow[e] * lds_eb2[e * Dm + d];
        size_t idx = (size_t)(t0 + m) * Dm + d;
        out[idx] = hidden[idx] + av * (oacc[mt][j][r] + bsum);
      }
    }
  }
}

// ---------------------------------------------------------------------------
extern "C" void kernel_launch(void* const* d_in, const int* in_sizes, int n_in,
                              void* d_out, int out_size, void* d_ws, size_t ws_size,
                              hipStream_t stream) {
  (void)in_sizes; (void)n_in; (void)out_size; (void)ws_size;
  const float* hidden = (const float*)d_in[0];
  const float* feat   = (const float*)d_in[1];
  const float* ln_g   = (const float*)d_in[2];
  const float* ln_b   = (const float*)d_in[3];
  const float* stageW = (const float*)d_in[4];
  const float* stageb = (const float*)d_in[5];
  const float* gfW    = (const float*)d_in[6];
  const float* gfb    = (const float*)d_in[7];
  const float* psW1   = (const float*)d_in[8];
  const float* psb1   = (const float*)d_in[9];
  const float* psW2   = (const float*)d_in[10];
  const float* psb2   = (const float*)d_in[11];
  const float* irW1   = (const float*)d_in[12];
  const float* irb1   = (const float*)d_in[13];
  const float* irW2   = (const float*)d_in[14];
  const float* irb2   = (const float*)d_in[15];
  const float* eW1    = (const float*)d_in[16];
  const float* eb1    = (const float*)d_in[17];
  const float* eW2    = (const float*)d_in[18];
  const float* eb2    = (const float*)d_in[19];
  const float* alpha  = (const float*)d_in[20];

  char* ws = (char*)d_ws;
  size_t off = 0;
  auto alloc = [&](size_t bytes) -> void* {
    void* p = ws + off;
    off += (bytes + 255) & ~(size_t)255;
    return p;
  };
  unsigned short* h_bf    = (unsigned short*)alloc((size_t)Tt * Dm * 2);
  unsigned short* stage_bf= (unsigned short*)alloc((size_t)Tt * DFe * 2);
  unsigned short* gemb_bf = (unsigned short*)alloc((size_t)Tt * Gg * DFe * 2);
  float* glog             = (float*)alloc((size_t)Tt * Gg * 4);
  float* ilogb            = (float*)alloc((size_t)Tt * Ee * 4);
  float* wcomb            = (float*)alloc((size_t)Tt * Ee * 4);
  unsigned* psW1p         = (unsigned*)alloc((size_t)(Gg * RIN / 2) * DRr * 4);
  unsigned* irW1p         = (unsigned*)alloc((size_t)(Gg * RIN / 2) * DRr * 4);
  unsigned* eW1p          = (unsigned*)alloc((size_t)(Ee * Dm / 2) * DHh * 4);
  unsigned* eW2p          = (unsigned*)alloc((size_t)(Ee * DHh / 2) * Dm * 4);

  // Fragment-native weight repacking.
  {
    int perblk = (RIN / 2) * DRr, tot = Gg * perblk;
    pack_b_frag<<<(tot + 255) / 256, 256, 0, stream>>>(psW1, psW1p, RIN, DRr, perblk, tot);
    pack_b_frag<<<(tot + 255) / 256, 256, 0, stream>>>(irW1, irW1p, RIN, DRr, perblk, tot);
  }
  {
    int perblk = (Dm / 2) * DHh, tot = Ee * perblk;
    pack_b_frag<<<(tot + 255) / 256, 256, 0, stream>>>(eW1, eW1p, Dm, DHh, perblk, tot);
  }
  {
    int perblk = (DHh / 2) * Dm, tot = Ee * perblk;
    pack_b_frag<<<(tot + 255) / 256, 256, 0, stream>>>(eW2, eW2p, DHh, Dm, perblk, tot);
  }

  prep_kernel<<<Tt, 256, 0, stream>>>(hidden, feat, ln_g, ln_b, stageW, stageb,
                                      gfW, gfb, h_bf, stage_bf, gemb_bf);

  dim3 rgrid(Tt / 16, Gg);
  router_kernel<<<rgrid, 256, 0, stream>>>(h_bf, stage_bf, gemb_bf,
                                           psW1p, psb1, psW2, psb2,
                                           irW1p, irb1, irW2, irb2,
                                           glog, ilogb);

  combine_kernel<<<Tt / 256, 256, 0, stream>>>(glog, ilogb, wcomb);

  expert_kernel<<<Tt / 32, 256, 0, stream>>>(h_bf, eW1p, eW2p, eb1, eb2,
                                             wcomb, hidden, alpha,
                                             (float*)d_out);
}
